// simVectorQuantizer_48155173323513
// MI455X (gfx1250) — matrix operand order repaired and checked by reading, not compiled, not run on hardware
//
#include <hip/hip_runtime.h>
#include <hip/hip_bf16.h>

// ---------------------------------------------------------------------------
// VQ quantizer, MI455X (gfx1250, wave32, WMMA + TDM).
// Dominant GEMM: (8192 x 8192 x 256) f32 via bf16 hi/lo split on
// v_wmma_f32_16x16x32_bf16. W tiles are staged into LDS by the Tensor Data
// Mover (tensor_load_to_lds, TENSORcnt, double buffered, TDM padding for
// bank-conflict-free fragment reads) and shared by all waves of the block.
// Softmax/argmin are fused into the tile loop (online stats) so the 256MB
// dot matrix is never materialized.
// ---------------------------------------------------------------------------

typedef __bf16 bfloat16_t;
typedef bfloat16_t v16bf __attribute__((ext_vector_type(16)));
typedef bfloat16_t v8bf  __attribute__((ext_vector_type(8)));
typedef float      v8f   __attribute__((ext_vector_type(8)));
typedef unsigned int v4u __attribute__((ext_vector_type(4)));
typedef int          v8i __attribute__((ext_vector_type(8)));
typedef int          v4i __attribute__((ext_vector_type(4)));

#define VQ_B 8
#define VQ_L 1024
#define VQ_D 256
#define VQ_K 8192
#define VQ_N (VQ_B * VQ_L)          // 8192 rows
#define INV_TEMP 100.0f             // 1/0.01
#define LROW 264                    // LDS row stride in bf16: 256 + 8 (TDM pad: 4dw/128dw)

// d_out layout (float32), tuple order of the reference return:
#define ZQ_OFF  0
#define CB_OFF  2097152
#define CM_OFF  2097153
#define EN_OFF  2097154
#define IDX_OFF 2097155
#define AP_OFF  2105347
#define MP_OFF  2105348
#define COS_OFF 2105349

// workspace layout (float offsets)
#define WS_W    0           // 2097152 f32  projected codebook W[K][D]
#define WS_W2   2097152     // 8192 f32     |W_k|^2
#define WS_Z2   2105344     // 8192 f32     |z_n|^2
#define WS_M1   2113536     // 8192 f32     per-row tempered softmax max
#define WS_Z1   2121728     // 8192 f32     per-row tempered softmax Z
#define WS_IDX  2129920     // 8192 i32     argmin index
#define WS_AVG  2138112     // 8192 f32     sum over rows of tempered probs
#define WS_SCAL 2146304     // 8 f32        [ent_sum, maxp_sum, sq_sum, cos_sum]
#define WS_WHI  2146320     // 2097152 bf16, 32B aligned
#define WS_WLO  (2146320 + 1048576)

// ---------------- online softmax helpers ------------------------------------
__device__ __forceinline__ void smax_upd3(float& mx, float& Z, float& S, float v) {
  if (v <= mx) { float e = __expf(v - mx); Z += e; S += (v - mx) * e; }
  else         { float c = __expf(mx - v); S = c * (S + (mx - v) * Z); Z = c * Z + 1.0f; mx = v; }
}
__device__ __forceinline__ void smax_upd2(float& mx, float& Z, float v) {
  if (v <= mx) Z += __expf(v - mx);
  else         { Z = __expf(mx - v) * Z + 1.0f; mx = v; }
}
__device__ __forceinline__ void smax_mrg3(float& mx, float& Z, float& S,
                                          float omx, float oZ, float oS) {
  if (omx <= mx) { float c = __expf(omx - mx); Z += c * oZ; S += c * (oS + (omx - mx) * oZ); }
  else           { float c = __expf(mx - omx); S = oS + c * (S + (mx - omx) * Z); Z = oZ + c * Z; mx = omx; }
}
__device__ __forceinline__ void smax_mrg2(float& mx, float& Z, float omx, float oZ) {
  if (omx <= mx) Z += __expf(omx - mx) * oZ;
  else           { Z = oZ + __expf(mx - omx) * Z; mx = omx; }
}

// ---------------- TDM: stage one 32x256 bf16 tile (row-padded) into LDS -----
// D# per CDNA5 ISA 8.3/8.4: 2D tile, data_size=2B, pad 4 dwords every 128
// dwords -> LDS row stride 528B (=LROW bf16), groups 2/3 unused (zero).
// amdgpu-toolchain builtin: 6 args (g0, g1, g2, g3, g4, cpol).
__device__ __forceinline__ void tdm_stage_tile(const bfloat16_t* gsrc, unsigned lds_addr) {
  const unsigned long long ga = (unsigned long long)(uintptr_t)gsrc;
  v4u g0;
  g0[0] = 1u;                                   // count=1, user descriptor
  g0[1] = lds_addr;                             // LDS byte address
  g0[2] = (unsigned)(ga & 0xFFFFFFFFu);         // global_addr[31:0]
  g0[3] = (unsigned)((ga >> 32) & 0x01FFFFFFu)  // global_addr[56:32]
          | 0x80000000u;                        // type=2 ("image")
  v8i g1;
  g1[0] = (int)0x07910000u;   // data_size=2B | pad_enable | interval=128dw | amount=4dw
  g1[1] = (int)(256u << 16);  // tensor_dim0 = 256 (row length, elements)
  g1[2] = (int)(32u  << 16);  // tensor_dim1 = 32 rows
  g1[3] = (int)(256u << 16);  // tile_dim0   = 256
  g1[4] = 32;                 // tile_dim1   = 32
  g1[5] = 256;                // tensor_dim0_stride = 256 elements
  g1[6] = 0;
  g1[7] = 0;
  v4i zz4 = {0, 0, 0, 0};
  v8i zz8 = {0, 0, 0, 0, 0, 0, 0, 0};
  __builtin_amdgcn_tensor_load_to_lds(g0, g1, zz4, zz4, zz8, 0);
}

// ---------------- WMMA 16x16 dot tile from LDS-staged fragments -------------
// Afh/Afl: 8 preloaded A fragments (registers). sBh_t/sBl_t: LDS tile base
// (16 codes x LROW). Result: lane l holds rows m=r+8*(l>=16), col n=base+(l&15).
__device__ __forceinline__ v8f dot_tile_lds(const v16bf* Afh, const v16bf* Afl,
                                            const bfloat16_t* sBh_t,
                                            const bfloat16_t* sBl_t, int lane) {
  v8f acc = {};
  const int half = lane >> 4, l16 = lane & 15;
  const bfloat16_t* brh = sBh_t + l16 * LROW;
  const bfloat16_t* brl = sBl_t + l16 * LROW;
#pragma unroll
  for (int c = 0; c < 8; ++c) {
    const int cb = c * 32 + half * 8;    // B frag: lane col n=l16, K in {cb..cb+7, cb+16..cb+23}
    union { v16bf v; v8bf h[2]; } Bh, Bl;
    Bh.h[0] = *(const v8bf*)(brh + cb);
    Bh.h[1] = *(const v8bf*)(brh + cb + 16);
    Bl.h[0] = *(const v8bf*)(brl + cb);
    Bl.h[1] = *(const v8bf*)(brl + cb + 16);
    acc = __builtin_amdgcn_wmma_f32_16x16x32_bf16(false, Afh[c], false, Bh.v, (short)0, acc, false, false);
    acc = __builtin_amdgcn_wmma_f32_16x16x32_bf16(false, Afh[c], false, Bl.v, (short)0, acc, false, false);
    acc = __builtin_amdgcn_wmma_f32_16x16x32_bf16(false, Afl[c], false, Bh.v, (short)0, acc, false, false);
  }
  return acc;
}

// ---------------- kernel: zero accumulators ---------------------------------
__global__ void vq_init(float* avg_acc, float* scal) {
  int i = blockIdx.x * blockDim.x + threadIdx.x;
  if (i < VQ_K) avg_acc[i] = 0.0f;
  if (i < 8)    scal[i]    = 0.0f;
}

// ---------------- kernel: W = emb @ projW^T + b -----------------------------
__global__ __launch_bounds__(256) void vq_project(const float* __restrict__ emb,
                                                  const float* __restrict__ pw,
                                                  const float* __restrict__ pb,
                                                  float* __restrict__ W) {
  __shared__ float se[VQ_D];
  const int k = blockIdx.x, j = threadIdx.x;
  se[j] = emb[(size_t)k * VQ_D + j];
  __syncthreads();
  float acc = pb[j];
  const float* pr = pw + (size_t)j * VQ_D;
#pragma unroll 8
  for (int d = 0; d < VQ_D; ++d) acc = fmaf(se[d], pr[d], acc);
  W[(size_t)k * VQ_D + j] = acc;
}

// ---------------- kernel: bf16 hi/lo split of W + row norms -----------------
__global__ __launch_bounds__(256) void vq_prep_w(const float* __restrict__ W,
                                                 bfloat16_t* __restrict__ Whi,
                                                 bfloat16_t* __restrict__ Wlo,
                                                 float* __restrict__ w2) {
  __shared__ float sm[VQ_D];
  const int k = blockIdx.x, d = threadIdx.x;
  float x = W[(size_t)k * VQ_D + d];
  bfloat16_t h = (bfloat16_t)x;
  bfloat16_t l = (bfloat16_t)(x - (float)h);
  Whi[(size_t)k * VQ_D + d] = h;
  Wlo[(size_t)k * VQ_D + d] = l;
  sm[d] = x * x;
  __syncthreads();
  for (int st = 128; st > 0; st >>= 1) { if (d < st) sm[d] += sm[d + st]; __syncthreads(); }
  if (d == 0) w2[k] = sm[0];
}

// ---------------- kernel: |z_n|^2 -------------------------------------------
__global__ __launch_bounds__(256) void vq_prep_z(const float* __restrict__ z,
                                                 float* __restrict__ z2) {
  __shared__ float sm[VQ_D];
  const int n = blockIdx.x, d = threadIdx.x;
  float x = z[(size_t)n * VQ_D + d];
  sm[d] = x * x;
  __syncthreads();
  for (int st = 128; st > 0; st >>= 1) { if (d < st) sm[d] += sm[d + st]; __syncthreads(); }
  if (d == 0) z2[n] = sm[0];
}

// ---------------- A staging + fragment preload (shared by both passes) ------
__device__ __forceinline__ void stage_A(const float* z, bfloat16_t (*sAh)[LROW],
                                        bfloat16_t (*sAl)[LROW], int rowbase, int lane) {
  const int r = lane >> 1, d0 = (lane & 1) * 128;
  const float* zr = z + (size_t)(rowbase + r) * VQ_D + d0;
  for (int d = 0; d < 128; ++d) {
    float x = zr[d];
    bfloat16_t h = (bfloat16_t)x;
    sAh[r][d0 + d] = h;
    sAl[r][d0 + d] = (bfloat16_t)(x - (float)h);
  }
}
__device__ __forceinline__ void load_A_frags(const bfloat16_t (*sAh)[LROW],
                                             const bfloat16_t (*sAl)[LROW],
                                             v16bf* Afh, v16bf* Afl, int lane) {
  const int half = lane >> 4, l16 = lane & 15;
  const bfloat16_t* arh = &sAh[l16][0];
  const bfloat16_t* arl = &sAl[l16][0];
#pragma unroll
  for (int c = 0; c < 8; ++c) {
    const int ca = c * 32 + half * 8;   // A frag: row m=l16, K in {ca..ca+7, ca+16..ca+23}
    union { v16bf v; v8bf h[2]; } th, tl;
    th.h[0] = *(const v8bf*)(arh + ca);
    th.h[1] = *(const v8bf*)(arh + ca + 16);
    tl.h[0] = *(const v8bf*)(arl + ca);
    tl.h[1] = *(const v8bf*)(arl + ca + 16);
    Afh[c] = th.v;
    Afl[c] = tl.v;
  }
}

// ---------------- pass A: argmin + online softmax stats ---------------------
#define WAVES_A 4
#define NSTAGE (VQ_K / 32)
__global__ __launch_bounds__(WAVES_A * 32) void vq_passA(
    const float* __restrict__ z, const bfloat16_t* __restrict__ Whi,
    const bfloat16_t* __restrict__ Wlo, const float* __restrict__ z2,
    const float* __restrict__ w2, float* __restrict__ out_idx_f,
    int* __restrict__ ws_idx, float* __restrict__ ws_m1, float* __restrict__ ws_Z1,
    float* __restrict__ scal) {
  __shared__ __align__(16) bfloat16_t sAh[WAVES_A][16][LROW];
  __shared__ __align__(16) bfloat16_t sAl[WAVES_A][16][LROW];
  __shared__ __align__(16) bfloat16_t sBh[2][32][LROW];
  __shared__ __align__(16) bfloat16_t sBl[2][32][LROW];
  const int tid = threadIdx.x;
  const int w = tid >> 5, lane = tid & 31;
  const int half = lane >> 4, l16 = lane & 15;
  const int rowbase = blockIdx.x * (WAVES_A * 16) + w * 16;

  if (tid == 0) {  // TDM prefetch of stage 0 (wave 0 issues, EXEC ignored)
    tdm_stage_tile(Whi, (unsigned)(uintptr_t)&sBh[0][0][0]);
    tdm_stage_tile(Wlo, (unsigned)(uintptr_t)&sBl[0][0][0]);
  }
  stage_A(z, sAh[w], sAl[w], rowbase, lane);
  __syncthreads();

  v16bf Afh[8], Afl[8];
  load_A_frags(sAh[w], sAl[w], Afh, Afl, lane);

  const int mbase = rowbase + 8 * half;
  float z2v[8];
#pragma unroll
  for (int r = 0; r < 8; ++r) z2v[r] = z2[mbase + r];

  float bd[8], mx1[8], Z1[8], S1[8], mx2[8], Z2[8];
  int bi[8];
#pragma unroll
  for (int r = 0; r < 8; ++r) {
    bd[r] = 3.4e38f; bi[r] = 0;
    mx1[r] = -3.4e38f; Z1[r] = 0.0f; S1[r] = 0.0f;
    mx2[r] = -3.4e38f; Z2[r] = 0.0f;
  }

  for (int s = 0; s < NSTAGE; ++s) {
    const int cur = s & 1;
    if (tid == 0) {
      if (s + 1 < NSTAGE) {  // prefetch next stage, wait for current
        tdm_stage_tile(Whi + (size_t)(s + 1) * 32 * VQ_D,
                       (unsigned)(uintptr_t)&sBh[cur ^ 1][0][0]);
        tdm_stage_tile(Wlo + (size_t)(s + 1) * 32 * VQ_D,
                       (unsigned)(uintptr_t)&sBl[cur ^ 1][0][0]);
        __builtin_amdgcn_s_wait_tensorcnt((short)2);
      } else {
        __builtin_amdgcn_s_wait_tensorcnt((short)0);
      }
    }
    __syncthreads();
#pragma unroll
    for (int t = 0; t < 2; ++t) {
      const int colbase = s * 32 + t * 16;
      v8f acc = dot_tile_lds(Afh, Afl, &sBh[cur][t * 16][0], &sBl[cur][t * 16][0], lane);
      const int   ncol = colbase + l16;
      const float w2n  = w2[ncol];
#pragma unroll
      for (int r = 0; r < 8; ++r) {
        const float sv = acc[r];
        const float dist = z2v[r] + w2n - 2.0f * sv;
        if (dist < bd[r]) { bd[r] = dist; bi[r] = ncol; }
        smax_upd3(mx1[r], Z1[r], S1[r], INV_TEMP * sv);  // tempered (entropy)
        smax_upd2(mx2[r], Z2[r], sv);                    // untempered (max prob)
      }
    }
    __syncthreads();  // all waves done reading before next TDM overwrite
  }

  // reduce across the 16 lanes of each half (they hold the same 8 rows)
  for (int off = 1; off <= 8; off <<= 1) {
#pragma unroll
    for (int r = 0; r < 8; ++r) {
      float obd = __shfl_xor(bd[r], off, 32);
      int   obi = __shfl_xor(bi[r], off, 32);
      if (obd < bd[r] || (obd == bd[r] && obi < bi[r])) { bd[r] = obd; bi[r] = obi; }
      float a = __shfl_xor(mx1[r], off, 32);
      float b = __shfl_xor(Z1[r],  off, 32);
      float c = __shfl_xor(S1[r],  off, 32);
      smax_mrg3(mx1[r], Z1[r], S1[r], a, b, c);
      float d = __shfl_xor(mx2[r], off, 32);
      float e = __shfl_xor(Z2[r],  off, 32);
      smax_mrg2(mx2[r], Z2[r], d, e);
    }
  }

  if (l16 == 0) {  // lanes 0 and 16 write their 8 rows
    float entloc = 0.0f, mploc = 0.0f;
#pragma unroll
    for (int r = 0; r < 8; ++r) {
      const int row = mbase + r;
      ws_idx[row]    = bi[r];
      out_idx_f[row] = (float)bi[r];
      ws_m1[row]     = mx1[r];
      ws_Z1[row]     = Z1[r];
      entloc += __logf(Z1[r]) - S1[r] / Z1[r];
      mploc  += 1.0f / Z2[r];
    }
    atomicAdd(&scal[0], entloc);
    atomicAdd(&scal[1], mploc);
  }
}

// ---------------- pass B: accumulate tempered probs per code ----------------
#define WAVES_B 2
__global__ __launch_bounds__(WAVES_B * 32) void vq_passB(
    const float* __restrict__ z, const bfloat16_t* __restrict__ Whi,
    const bfloat16_t* __restrict__ Wlo, const float* __restrict__ ws_m1,
    const float* __restrict__ ws_Z1, float* __restrict__ avg_acc) {
  __shared__ __align__(16) bfloat16_t sAh[WAVES_B][16][LROW];
  __shared__ __align__(16) bfloat16_t sAl[WAVES_B][16][LROW];
  __shared__ __align__(16) bfloat16_t sBh[2][32][LROW];
  __shared__ __align__(16) bfloat16_t sBl[2][32][LROW];
  __shared__ float accl[VQ_K];
  const int tid = threadIdx.x;
  const int w = tid >> 5, lane = tid & 31;
  const int half = lane >> 4, l16 = lane & 15;
  const int rowbase = blockIdx.x * (WAVES_B * 16) + w * 16;

  if (tid == 0) {
    tdm_stage_tile(Whi, (unsigned)(uintptr_t)&sBh[0][0][0]);
    tdm_stage_tile(Wlo, (unsigned)(uintptr_t)&sBl[0][0][0]);
  }
  for (int i = tid; i < VQ_K; i += WAVES_B * 32) accl[i] = 0.0f;
  stage_A(z, sAh[w], sAl[w], rowbase, lane);
  __syncthreads();

  v16bf Afh[8], Afl[8];
  load_A_frags(sAh[w], sAl[w], Afh, Afl, lane);

  const int mbase = rowbase + 8 * half;
  float m1v[8], izv[8];
#pragma unroll
  for (int r = 0; r < 8; ++r) {
    m1v[r] = ws_m1[mbase + r];
    izv[r] = 1.0f / ws_Z1[mbase + r];
  }

  for (int s = 0; s < NSTAGE; ++s) {
    const int cur = s & 1;
    if (tid == 0) {
      if (s + 1 < NSTAGE) {
        tdm_stage_tile(Whi + (size_t)(s + 1) * 32 * VQ_D,
                       (unsigned)(uintptr_t)&sBh[cur ^ 1][0][0]);
        tdm_stage_tile(Wlo + (size_t)(s + 1) * 32 * VQ_D,
                       (unsigned)(uintptr_t)&sBl[cur ^ 1][0][0]);
        __builtin_amdgcn_s_wait_tensorcnt((short)2);
      } else {
        __builtin_amdgcn_s_wait_tensorcnt((short)0);
      }
    }
    __syncthreads();
#pragma unroll
    for (int t = 0; t < 2; ++t) {
      const int colbase = s * 32 + t * 16;
      v8f acc = dot_tile_lds(Afh, Afl, &sBh[cur][t * 16][0], &sBl[cur][t * 16][0], lane);
      float csum = 0.0f;
#pragma unroll
      for (int r = 0; r < 8; ++r)
        csum += __expf(INV_TEMP * acc[r] - m1v[r]) * izv[r];
      csum += __shfl_xor(csum, 16, 32);  // combine both 8-row halves
      if (lane < 16) atomicAdd(&accl[colbase + lane], csum);
    }
    __syncthreads();
  }
  for (int i = tid; i < VQ_K; i += WAVES_B * 32) atomicAdd(&avg_acc[i], accl[i]);
}

// ---------------- finalize rows: z_q = W[idx], losses, cosine ---------------
__global__ __launch_bounds__(256) void vq_finalize_rows(
    const float* __restrict__ z, const float* __restrict__ W,
    const int* __restrict__ ws_idx, float* __restrict__ zq, float* __restrict__ scal) {
  __shared__ float sm[VQ_D];
  const int row = blockIdx.x, d = threadIdx.x;
  const int idx = ws_idx[row];
  const float zv = z[(size_t)row * VQ_D + d];
  const float qv = W[(size_t)idx * VQ_D + d];
  zq[(size_t)row * VQ_D + d] = qv;
  const float diff = qv - zv;

  float red[4] = { diff * diff, zv * qv, zv * zv, qv * qv };
  float res[4];
#pragma unroll
  for (int i = 0; i < 4; ++i) {
    sm[d] = red[i];
    __syncthreads();
    for (int st = 128; st > 0; st >>= 1) { if (d < st) sm[d] += sm[d + st]; __syncthreads(); }
    res[i] = sm[0];
    __syncthreads();
  }
  if (d == 0) {
    atomicAdd(&scal[2], res[0]);
    float cosr = res[1] / (fmaxf(sqrtf(res[2]), 1e-8f) * fmaxf(sqrtf(res[3]), 1e-8f));
    atomicAdd(&scal[3], cosr);
  }
}

// ---------------- final scalars ---------------------------------------------
__global__ __launch_bounds__(256) void vq_final(const float* __restrict__ avg_acc,
                                                const float* __restrict__ scal,
                                                float* __restrict__ out) {
  __shared__ float sm[256];
  const int t = threadIdx.x;
  float local = 0.0f;
  for (int k = t; k < VQ_K; k += 256) {
    float a = avg_acc[k] * (1.0f / (float)VQ_N);
    local += a * __logf(a + 1e-6f);
  }
  sm[t] = local;
  __syncthreads();
  for (int st = 128; st > 0; st >>= 1) { if (t < st) sm[t] += sm[t + st]; __syncthreads(); }
  if (t == 0) {
    const float avg_entropy    = -sm[0];
    const float sample_entropy = scal[0] / (float)VQ_N;
    const float mse            = scal[2] / (float)(VQ_N * VQ_D);
    out[EN_OFF]  = 0.1f * (sample_entropy - avg_entropy);
    out[CB_OFF]  = mse;
    out[CM_OFF]  = 0.25f * mse;
    out[AP_OFF]  = 1.0f / (float)VQ_K;   // mean of softmax == 1/K exactly
    out[MP_OFF]  = scal[1] / (float)VQ_N;
    out[COS_OFF] = scal[3] / (float)VQ_N;
  }
}

// ---------------------------------------------------------------------------
extern "C" void kernel_launch(void* const* d_in, const int* in_sizes, int n_in,
                              void* d_out, int out_size, void* d_ws, size_t ws_size,
                              hipStream_t stream) {
  const float* z   = (const float*)d_in[0];   // (8,1024,256)
  const float* emb = (const float*)d_in[1];   // (8192,256)
  const float* pw  = (const float*)d_in[2];   // (256,256)
  const float* pb  = (const float*)d_in[3];   // (256,)
  float* out = (float*)d_out;

  float* wsf       = (float*)d_ws;            // needs ~16.2 MiB workspace
  float* W         = wsf + WS_W;
  float* w2        = wsf + WS_W2;
  float* z2        = wsf + WS_Z2;
  float* m1        = wsf + WS_M1;
  float* Z1        = wsf + WS_Z1;
  int*   idxi      = (int*)(wsf + WS_IDX);
  float* avg_acc   = wsf + WS_AVG;
  float* scal      = wsf + WS_SCAL;
  bfloat16_t* Whi  = (bfloat16_t*)(wsf + WS_WHI);
  bfloat16_t* Wlo  = (bfloat16_t*)(wsf + WS_WLO);

  vq_init<<<32, 256, 0, stream>>>(avg_acc, scal);
  vq_project<<<VQ_K, 256, 0, stream>>>(emb, pw, pb, W);
  vq_prep_w<<<VQ_K, 256, 0, stream>>>(W, Whi, Wlo, w2);
  vq_prep_z<<<VQ_N, 256, 0, stream>>>(z, z2);
  vq_passA<<<VQ_N / (WAVES_A * 16), WAVES_A * 32, 0, stream>>>(
      z, Whi, Wlo, z2, w2, out + IDX_OFF, idxi, m1, Z1, scal);
  vq_passB<<<VQ_N / (WAVES_B * 16), WAVES_B * 32, 0, stream>>>(
      z, Whi, Wlo, m1, Z1, avg_acc);
  vq_finalize_rows<<<VQ_N, 256, 0, stream>>>(z, W, idxi, out + ZQ_OFF, scal);
  vq_final<<<1, 256, 0, stream>>>(avg_acc, scal, out);
}
